// MoELayer_46291157516841
// MI455X (gfx1250) — matrix operand order repaired and checked by reading, not compiled, not run on hardware
//
#include <hip/hip_runtime.h>

typedef __attribute__((ext_vector_type(16))) _Float16 v16h;
typedef __attribute__((ext_vector_type(8)))  float    v8f;
typedef unsigned short u16t;

#define C_DIM 768
#define F_DIM 3072
#define NEXP  8
#define NTOK  16384
#define MT    64            // token rows per block tile
#define FT    128           // F chunk
#define XS_STR  (C_DIM + 8) // 776 (pad to break bank conflicts)
#define W1S_STR (C_DIM + 8) // transposed W1 chunk: [FT][C]
#define W2S_STR (FT + 8)    // transposed W2 chunk: [C][FT]
#define HS_STR  (FT + 8)
#define WS_ELTS (C_DIM * W2S_STR)   // 104448 >= FT*W1S_STR (99328)

union FragU { v16h h; uint4 q[2]; };

__device__ __forceinline__ u16t f2h(float f) {
  union { _Float16 h; u16t u; } cv;
  cv.h = (_Float16)f;
  return cv.u;
}

__device__ __forceinline__ v8f wmma_f16(v16h a, v16h b, v8f c) {
  return __builtin_amdgcn_wmma_f32_16x16x32_f16(false, a, false, b, (short)0, c,
                                                false, false);
}

// ---------------- init ----------------
__global__ void moe_zero_kernel(float4* __restrict__ out4, int n4,
                                int* __restrict__ cnt, float* __restrict__ psum) {
  int i = blockIdx.x * blockDim.x + threadIdx.x;
  if (i < n4) out4[i] = make_float4(0.f, 0.f, 0.f, 0.f);
  if (blockIdx.x == 0 && threadIdx.x < NEXP) {
    cnt[threadIdx.x] = 0;
    psum[threadIdx.x] = 0.f;
  }
}

// ---------------- fp32 -> fp16 ----------------
__global__ void moe_cvt_kernel(const float4* __restrict__ src,
                               ushort4* __restrict__ dst, int n4) {
  int i = blockIdx.x * blockDim.x + threadIdx.x;
  if (i < n4) {
    float4 v = src[i];
    ushort4 h;
    h.x = f2h(v.x); h.y = f2h(v.y); h.z = f2h(v.z); h.w = f2h(v.w);
    dst[i] = h;
  }
}

// ---------------- router: one wave32 per token ----------------
__global__ __launch_bounds__(256) void moe_router_kernel(
    const float* __restrict__ x, const float* __restrict__ rw,
    int* __restrict__ bidx, float* __restrict__ bgate,
    int* __restrict__ cnt, float* __restrict__ psum) {
  const int wave = threadIdx.x >> 5;
  const int lane = threadIdx.x & 31;
  const int token = blockIdx.x * 8 + wave;

  float p[NEXP];
#pragma unroll
  for (int e = 0; e < NEXP; ++e) p[e] = 0.f;

  for (int c = lane; c < C_DIM; c += 32) {
    float xv = x[(size_t)token * C_DIM + c];
#pragma unroll
    for (int e = 0; e < NEXP; ++e) p[e] += xv * rw[e * C_DIM + c];
  }
#pragma unroll
  for (int e = 0; e < NEXP; ++e)
    for (int off = 16; off > 0; off >>= 1)
      p[e] += __shfl_xor(p[e], off, 32);

  if (lane == 0) {
    float m = p[0];
#pragma unroll
    for (int e = 1; e < NEXP; ++e) m = fmaxf(m, p[e]);
    float pr[NEXP];
    float s = 0.f;
#pragma unroll
    for (int e = 0; e < NEXP; ++e) { pr[e] = __expf(p[e] - m); s += pr[e]; }
    float inv = 1.f / s;
#pragma unroll
    for (int e = 0; e < NEXP; ++e) pr[e] *= inv;

    int i1 = 0;
#pragma unroll
    for (int e = 1; e < NEXP; ++e) if (pr[e] > pr[i1]) i1 = e;
    int i2 = (i1 == 0) ? 1 : 0;
#pragma unroll
    for (int e = 0; e < NEXP; ++e) if (e != i1 && pr[e] > pr[i2]) i2 = e;

    float g1 = pr[i1], g2 = pr[i2];
    float gs = 1.f / (g1 + g2);
    g1 *= gs; g2 *= gs;

#pragma unroll
    for (int e = 0; e < NEXP; ++e) atomicAdd(&psum[e], pr[e]);

    int p1 = atomicAdd(&cnt[i1], 1);
    bidx[i1 * NTOK + p1] = token;
    bgate[i1 * NTOK + p1] = g1;
    int p2 = atomicAdd(&cnt[i2], 1);
    bidx[i2 * NTOK + p2] = token;
    bgate[i2 * NTOK + p2] = g2;
  }
}

// ---------------- aux loss ----------------
__global__ void moe_aux_kernel(const int* __restrict__ cnt,
                               const float* __restrict__ psum,
                               float* __restrict__ out_aux) {
  if (threadIdx.x == 0 && blockIdx.x == 0) {
    float s = 0.f;
#pragma unroll
    for (int e = 0; e < NEXP; ++e) s += (float)cnt[e] * psum[e];
    out_aux[0] = (float)NEXP * s / ((float)NTOK * (float)NTOK);
  }
}

// ---------------- fused routed FFN (WMMA + async-to-LDS) ----------------
__global__ __launch_bounds__(1024) void moe_ffn_kernel(
    const u16t* __restrict__ xh,    // [NTOK, C] f16
    const u16t* __restrict__ w1h,   // [E, C, F] f16
    const float* __restrict__ b1,   // [E, F]
    const u16t* __restrict__ w2h,   // [E, F, C] f16
    const float* __restrict__ b2,   // [E, C]
    const int* __restrict__ bidx,   // [E, NTOK]
    const float* __restrict__ bgate,// [E, NTOK]
    const int* __restrict__ cnt,    // [E]
    float* __restrict__ out) {      // [NTOK, C]
  __shared__ alignas(16) u16t Xs[MT * XS_STR];  // x tile, f16, [m][c]
  __shared__ alignas(16) u16t Ws[WS_ELTS];      // transposed W1 or W2 chunk
  __shared__ alignas(16) u16t Hs[MT * HS_STR];  // gelu(h) tile, [m][fl]
  __shared__ int   toks[MT];
  __shared__ float gats[MT];

  const int e    = blockIdx.x >> 8;   // 256 tiles per expert
  const int tile = blockIdx.x & 255;
  const int ce   = cnt[e];
  if (tile * MT >= ce) return;

  const int tid = threadIdx.x;
  if (tid < MT) {
    int m = tile * MT + tid;
    if (m < ce) { toks[tid] = bidx[e * NTOK + m]; gats[tid] = bgate[e * NTOK + m]; }
    else        { toks[tid] = 0;                  gats[tid] = 0.0f; }
  }
  __syncthreads();

  // stage X tile with CDNA5 async copy: f16 rows gathered straight into LDS,
  // tracked on ASYNCcnt (no VGPR round trip).
  for (int i = tid; i < MT * (C_DIM / 8); i += 1024) {   // 6144 x 16B copies
    int m  = i / (C_DIM / 8);
    int c8 = (i - m * (C_DIM / 8)) * 8;
    unsigned long long gaddr =
        (unsigned long long)(size_t)(xh + (size_t)toks[m] * C_DIM + c8);
    unsigned lds_off = (unsigned)(size_t)(&Xs[m * XS_STR + c8]);
    asm volatile("global_load_async_to_lds_b128 %0, %1, off"
                 :: "v"(lds_off), "v"(gaddr) : "memory");
  }
  asm volatile("s_wait_asynccnt 0x0" ::: "memory");

  const int wave = tid >> 5;
  const int lane = tid & 31;
  const int ln   = lane & 15;
  const int sel  = lane >> 4;
  const int mi = wave >> 3, fj = wave & 7;  // GEMM1: tile (mi, fj) of 64x128 H
  const int r2 = wave >> 3, g2 = wave & 7;  // GEMM2: row block r2, col group g2

  v8f acc[6];  // 6 persistent 16x16 f32 tiles of out[64 x 768]
#pragma unroll
  for (int t = 0; t < 6; ++t) {
    float bv = b2[e * C_DIM + (g2 + 8 * t) * 16 + ln];
#pragma unroll
    for (int i = 0; i < 8; ++i) acc[t][i] = bv;
  }

  for (int fc = 0; fc < F_DIM; fc += FT) {
    __syncthreads();  // prior GEMM2 done reading Ws / X staging visible
    // stage W1 chunk transposed: Ws[fl][c] = w1[e][c][fc+fl]
    {
      const u16t* g = w1h + (size_t)e * C_DIM * F_DIM + fc;
      for (int i = tid; i < C_DIM * (FT / 8); i += 1024) {
        int c  = i >> 4;
        int f0 = (i & 15) * 8;
        union { uint4 q; u16t s[8]; } u;
        u.q = *(const uint4*)(g + (size_t)c * F_DIM + f0);
        if ((i & 7) == 0 && fc + FT < F_DIM)  // pull next chunk toward WGP
          __builtin_prefetch(g + (size_t)c * F_DIM + f0 + FT, 0, 3);
#pragma unroll
        for (int j = 0; j < 8; ++j) Ws[(f0 + j) * W1S_STR + c] = u.s[j];
      }
    }
    __syncthreads();

    // GEMM1: H[mi,fj] = gelu(X @ W1chunk + b1)
    v8f hacc;
    {
      float bv = b1[e * F_DIM + fc + fj * 16 + ln];
#pragma unroll
      for (int i = 0; i < 8; ++i) hacc[i] = bv;
    }
    const u16t* xr = &Xs[(mi * 16 + ln) * XS_STR];
    const u16t* wr = &Ws[(fj * 16 + ln) * W1S_STR];
#pragma unroll 4
    for (int kk = 0; kk < C_DIM; kk += 32) {
      FragU A, B;
      A.q[0] = *(const uint4*)(xr + kk + sel * 8);
      A.q[1] = *(const uint4*)(xr + kk + 16 + sel * 8);
      B.q[0] = *(const uint4*)(wr + kk + sel * 8);
      B.q[1] = *(const uint4*)(wr + kk + 16 + sel * 8);
      hacc = wmma_f16(A.h, B.h, hacc);
    }
#pragma unroll
    for (int i = 0; i < 8; ++i) {
      float v  = hacc[i];
      float gl = 0.5f * v * (1.0f + erff(v * 0.70710678118f));  // exact gelu
      Hs[(mi * 16 + sel * 8 + i) * HS_STR + fj * 16 + ln] = f2h(gl);
    }
    __syncthreads();  // Hs visible; all waves done reading W1 chunk

    // stage W2 chunk transposed: Ws[c][fl] = w2[e][fc+fl][c]
    {
      const u16t* g = w2h + (size_t)e * F_DIM * C_DIM + (size_t)fc * C_DIM;
      for (int i = tid; i < FT * (C_DIM / 8); i += 1024) {
        int fl = i / (C_DIM / 8);
        int c0 = (i - fl * (C_DIM / 8)) * 8;
        union { uint4 q; u16t s[8]; } u;
        u.q = *(const uint4*)(g + (size_t)fl * C_DIM + c0);
        if ((i & 7) == 0 && fc + FT < F_DIM)
          __builtin_prefetch(g + (size_t)(fl + FT) * C_DIM + c0, 0, 3);
#pragma unroll
        for (int j = 0; j < 8; ++j) Ws[(c0 + j) * W2S_STR + fl] = u.s[j];
      }
    }
    __syncthreads();

    // GEMM2: out += H @ W2chunk (software-pipelined 24-step stream:
    // loads for step s+1 are issued before the WMMA of step s, so each
    // v_wmma waits on loads issued one step earlier, not on its own)
    {
      const u16t* hr = &Hs[(r2 * 16 + ln) * HS_STR];
      FragU Acur, Anxt, Bcur, Bnxt;
      Acur.q[0] = *(const uint4*)(hr + sel * 8);
      Acur.q[1] = *(const uint4*)(hr + 16 + sel * 8);
      {
        const int col0 = g2 * 16 + ln;
        Bcur.q[0] = *(const uint4*)&Ws[col0 * W2S_STR + sel * 8];
        Bcur.q[1] = *(const uint4*)&Ws[col0 * W2S_STR + 16 + sel * 8];
      }
#pragma unroll
      for (int s = 0; s < 24; ++s) {
        const int t  = s % 6;
        const int sn = s + 1;
        if (sn < 24) {
          const int tn   = sn % 6;
          const int kkn  = (sn / 6) * 32;
          const int coln = (g2 + 8 * tn) * 16 + ln;
          Bnxt.q[0] = *(const uint4*)&Ws[coln * W2S_STR + kkn + sel * 8];
          Bnxt.q[1] = *(const uint4*)&Ws[coln * W2S_STR + kkn + 16 + sel * 8];
          if (tn == 0) {
            Anxt.q[0] = *(const uint4*)(hr + kkn + sel * 8);
            Anxt.q[1] = *(const uint4*)(hr + kkn + 16 + sel * 8);
          }
        }
        acc[t] = wmma_f16(Acur.h, Bcur.h, acc[t]);
        if (sn < 24) {
          Bcur = Bnxt;
          if (t == 5) Acur = Anxt;
        }
      }
    }
  }

  // epilogue: gated atomic accumulate into out
  int   rt[8];
  float rg[8];
#pragma unroll
  for (int i = 0; i < 8; ++i) {
    int m = r2 * 16 + sel * 8 + i;
    rt[i] = toks[m];
    rg[i] = gats[m];
  }
#pragma unroll
  for (int t = 0; t < 6; ++t) {
    int col = (g2 + 8 * t) * 16 + ln;
#pragma unroll
    for (int i = 0; i < 8; ++i)
      atomicAdd(&out[(size_t)rt[i] * C_DIM + col], rg[i] * acc[t][i]);
  }
}

extern "C" void kernel_launch(void* const* d_in, const int* in_sizes, int n_in,
                              void* d_out, int out_size, void* d_ws, size_t ws_size,
                              hipStream_t stream) {
  (void)in_sizes; (void)n_in; (void)out_size; (void)ws_size;
  const float* x  = (const float*)d_in[0];
  const float* rw = (const float*)d_in[1];
  const float* w1 = (const float*)d_in[2];
  const float* b1 = (const float*)d_in[3];
  const float* w2 = (const float*)d_in[4];
  const float* b2 = (const float*)d_in[5];
  float* out = (float*)d_out;

  char* ws = (char*)d_ws;
  size_t o = 0;
  u16t* w1h = (u16t*)(ws + o); o += (size_t)NEXP * C_DIM * F_DIM * 2;
  u16t* w2h = (u16t*)(ws + o); o += (size_t)NEXP * C_DIM * F_DIM * 2;
  u16t* xhh = (u16t*)(ws + o); o += (size_t)NTOK * C_DIM * 2;
  int*   bidx  = (int*)(ws + o);   o += (size_t)NEXP * NTOK * 4;
  float* bgate = (float*)(ws + o); o += (size_t)NEXP * NTOK * 4;
  int*   cnt   = (int*)(ws + o);   o += 256;
  float* psum  = (float*)(ws + o); o += 256;

  const int n4_out = NTOK * C_DIM / 4;                // 3,145,728
  const int n4_w   = NEXP * C_DIM * F_DIM / 4;        // 4,718,592
  const int n4_x   = NTOK * C_DIM / 4;                // 3,145,728

  moe_zero_kernel<<<(n4_out + 255) / 256, 256, 0, stream>>>((float4*)out, n4_out,
                                                            cnt, psum);
  moe_cvt_kernel<<<(n4_w + 255) / 256, 256, 0, stream>>>((const float4*)w1,
                                                         (ushort4*)w1h, n4_w);
  moe_cvt_kernel<<<(n4_w + 255) / 256, 256, 0, stream>>>((const float4*)w2,
                                                         (ushort4*)w2h, n4_w);
  moe_cvt_kernel<<<(n4_x + 255) / 256, 256, 0, stream>>>((const float4*)x,
                                                         (ushort4*)xhh, n4_x);
  moe_router_kernel<<<NTOK / 8, 256, 0, stream>>>(x, rw, bidx, bgate, cnt, psum);
  moe_aux_kernel<<<1, 64, 0, stream>>>(cnt, psum, out + (size_t)NTOK * C_DIM);
  moe_ffn_kernel<<<NEXP * 256, 1024, 0, stream>>>(xhh, w1h, b1, w2h, b2, bidx,
                                                  bgate, cnt, out);
}
